// ResidualMambaBlockLayer_39049842655409
// MI455X (gfx1250) — matrix operand order repaired and checked by reading, not compiled, not run on hardware
//
#include <hip/hip_runtime.h>
#include <hip/hip_bf16.h>

#define BATCH   2
#define SEQ     2048
#define DMODEL  256
#define DSTATE  16
#define DCONV   4
#define DINNER  512
#define DTRANK  16
#define NTOK    (BATCH*SEQ)         /* 4096 tokens */
#define NDBL    (DTRANK + 2*DSTATE) /* 48 */

typedef __attribute__((ext_vector_type(16))) __bf16 v16bf;
typedef __attribute__((ext_vector_type(8)))  __bf16 v8bf;
typedef __attribute__((ext_vector_type(8)))  float  v8f;

__device__ __forceinline__ float sigmoidf_(float v){ return 1.f/(1.f+__expf(-v)); }
__device__ __forceinline__ float softplusf_(float v){ return v > 20.f ? v : log1pf(__expf(v)); }

// ---------------------------------------------------------------- f32 -> bf16
__global__ void f2bf_kernel(const float* __restrict__ s, __bf16* __restrict__ d, int n){
  int i = blockIdx.x*blockDim.x + threadIdx.x;
  if (i < n) d[i] = (__bf16)s[i];
}

// ------------------------------------------------- RMSNorm + transpose to (B*L, C)
// x: (B, C, 1, L) fp32 ; hbf: (B*L, C) bf16
__global__ void rmsnorm_kernel(const float* __restrict__ x, const float* __restrict__ w,
                               __bf16* __restrict__ hbf){
  __shared__ float red[DMODEL];
  int m = blockIdx.x;                 // token index b*SEQ + l
  int c = threadIdx.x;                // channel
  int b = m >> 11, l = m & (SEQ-1);
  float v = x[((size_t)(b*DMODEL + c))*SEQ + l];
  red[c] = v*v;
  __syncthreads();
  for (int off = DMODEL/2; off > 0; off >>= 1){
    if (c < off) red[c] += red[c+off];
    __syncthreads();
  }
  float scale = rsqrtf(red[0]*(1.f/DMODEL) + 1e-5f);
  hbf[(size_t)m*DMODEL + c] = (__bf16)(v*scale*w[c]);
}

// ------------------------------------------------- WMMA fragment loaders
// A fragment 16x32 bf16: lane holds row (lane&15); hi=lane>>4 selects K-halves.
__device__ __forceinline__ v16bf load_a_frag(const __bf16* __restrict__ A, int lda,
                                             int row, int k0, int hi){
  const __bf16* p = A + (size_t)row*lda + k0 + hi*16;
  v8bf lo = *(const v8bf*)p;          // K = k0 + hi*16 .. +7
  v8bf up = *(const v8bf*)(p + 8);    // K = k0 + hi*16 + 8 .. +15
  v16bf r;
  #pragma unroll
  for (int i=0;i<8;i++){ r[i] = lo[i]; r[8+i] = up[i]; }
  return r;
}
// B fragment 32x16 bf16 with W row-major over K (W[n][k]): 16 contiguous K per lane.
__device__ __forceinline__ v16bf load_b_frag(const __bf16* __restrict__ W, int ldw,
                                             int col, int k0, int hi){
  return *(const v16bf*)(W + (size_t)col*ldw + k0 + hi*16);
}
#define WMMA_BF16(a,b,c) __builtin_amdgcn_wmma_f32_16x16x32_bf16(false,(a),false,(b),(short)0,(c),false,false)

// ------------------------------------------------- generic GEMM  C[M,N] = A[M,K] * W[N,K]^T
// Wave computes a (16*MSUB) x (16*NSUB) tile; tile index scalarized via readfirstlane
// so EXEC is never touched around the WMMAs (straight-line unrolled inner loop).
template<int MSUB, int NSUB>
__global__ void gemm_bf16_t(const __bf16* __restrict__ A, const __bf16* __restrict__ W,
                            float* __restrict__ C,
                            int K, int lda, int ldw, int ldc,
                            int ntiles, int total_tiles){
  const int lane = threadIdx.x & 31;
  int tile = __builtin_amdgcn_readfirstlane(blockIdx.x*(blockDim.x >> 5) + (threadIdx.x >> 5));
  if (tile >= total_tiles) return;     // scalar branch
  const int mt = tile / ntiles, nt = tile % ntiles;
  const int m0 = mt*(16*MSUB), n0 = nt*(16*NSUB);
  const int hi = lane >> 4;
  const int lo16 = lane & 15;
  v8f acc[MSUB][NSUB];
  #pragma unroll
  for (int i=0;i<MSUB;i++)
    #pragma unroll
    for (int j=0;j<NSUB;j++) acc[i][j] = (v8f){};

  for (int k0 = 0; k0 < K; k0 += 32){
    v16bf a[MSUB], b[NSUB];
    #pragma unroll
    for (int i=0;i<MSUB;i++) a[i] = load_a_frag(A, lda, m0 + i*16 + lo16, k0, hi);
    #pragma unroll
    for (int j=0;j<NSUB;j++) b[j] = load_b_frag(W, ldw, n0 + j*16 + lo16, k0, hi);
    if (k0 + 32 < K){
      __builtin_prefetch(A + (size_t)(m0 + lo16)*lda + k0 + 32, 0, 1);
      __builtin_prefetch(W + (size_t)(n0 + lo16)*ldw + k0 + 32, 0, 1);
    }
    #pragma unroll
    for (int i=0;i<MSUB;i++)
      #pragma unroll
      for (int j=0;j<NSUB;j++) acc[i][j] = WMMA_BF16(a[i], b[j], acc[i][j]);
  }
  #pragma unroll
  for (int i=0;i<MSUB;i++){
    int rbase = m0 + i*16 + hi*8;
    #pragma unroll
    for (int r=0;r<8;r++){
      size_t rowoff = (size_t)(rbase + r)*ldc;
      #pragma unroll
      for (int j=0;j<NSUB;j++) C[rowoff + n0 + j*16 + lo16] = acc[i][j][r];
    }
  }
}

// ------------------------------------------------- out_proj GEMM + residual + transpose
// out[b,c,0,l] = sum_d Y[m,d]*Wout[c,d] + x[b,c,0,l],  m = b*SEQ + l
__global__ void gemm_out_residual_kernel(const __bf16* __restrict__ Y, const __bf16* __restrict__ W,
                                         const float* __restrict__ xres, float* __restrict__ out){
  const int MSUB = 2, NSUB = 4;
  const int ntiles = DMODEL/(16*NSUB);   // 4
  const int mtiles = NTOK/(16*MSUB);     // 128
  const int lane = threadIdx.x & 31;
  int tile = __builtin_amdgcn_readfirstlane(blockIdx.x*(blockDim.x >> 5) + (threadIdx.x >> 5));
  if (tile >= mtiles*ntiles) return;
  const int mt = tile / ntiles, nt = tile % ntiles;
  const int m0 = mt*(16*MSUB), n0 = nt*(16*NSUB);
  const int hi = lane >> 4;
  const int lo16 = lane & 15;
  v8f acc[MSUB][NSUB];
  #pragma unroll
  for (int i=0;i<MSUB;i++)
    #pragma unroll
    for (int j=0;j<NSUB;j++) acc[i][j] = (v8f){};

  for (int k0 = 0; k0 < DINNER; k0 += 32){
    v16bf a[MSUB], b[NSUB];
    #pragma unroll
    for (int i=0;i<MSUB;i++) a[i] = load_a_frag(Y, DINNER, m0 + i*16 + lo16, k0, hi);
    #pragma unroll
    for (int j=0;j<NSUB;j++) b[j] = load_b_frag(W, DINNER, n0 + j*16 + lo16, k0, hi);
    if (k0 + 32 < DINNER){
      __builtin_prefetch(Y + (size_t)(m0 + lo16)*DINNER + k0 + 32, 0, 1);
      __builtin_prefetch(W + (size_t)(n0 + lo16)*DINNER + k0 + 32, 0, 1);
    }
    #pragma unroll
    for (int i=0;i<MSUB;i++)
      #pragma unroll
      for (int j=0;j<NSUB;j++) acc[i][j] = WMMA_BF16(a[i], b[j], acc[i][j]);
  }
  #pragma unroll
  for (int i=0;i<MSUB;i++){
    int rbase = m0 + i*16 + hi*8;
    #pragma unroll
    for (int r=0;r<8;r++){
      int m = rbase + r;
      int b = m >> 11, l = m & (SEQ-1);
      size_t base = ((size_t)b*DMODEL)*SEQ + l;
      #pragma unroll
      for (int j=0;j<NSUB;j++){
        size_t o = base + (size_t)(n0 + j*16 + lo16)*SEQ;
        out[o] = acc[i][j][r] + xres[o];
      }
    }
  }
}

// ------------------------------------------------- depthwise causal conv1d + SiLU
// xz: (B*L, 2*DINNER) ; xin = xz[:, :DINNER]
__global__ void conv_silu_kernel(const float* __restrict__ xz, const float* __restrict__ cw,
                                 const float* __restrict__ cb,
                                 float* __restrict__ xc, __bf16* __restrict__ xcbf){
  int t = blockIdx.x*blockDim.x + threadIdx.x;
  if (t >= NTOK*DINNER) return;
  int d = t & (DINNER-1);
  int m = t >> 9;
  int l = m & (SEQ-1);
  float acc = cb[d];
  #pragma unroll
  for (int j = 0; j < DCONV; j++){
    int li = l - (DCONV-1) + j;
    if (li >= 0) acc += cw[d*DCONV + j] * xz[(size_t)(m - (DCONV-1) + j)*(2*DINNER) + d];
  }
  float v = acc * sigmoidf_(acc);
  xc[t]   = v;
  xcbf[t] = (__bf16)v;
}

// ------------------------------------------------- delta = softplus(dt @ Wdt^T + b)
__global__ void dtproj_kernel(const float* __restrict__ dbl, const float* __restrict__ wdt,
                              const float* __restrict__ bdt, float* __restrict__ delta){
  int t = blockIdx.x*blockDim.x + threadIdx.x;
  if (t >= NTOK*DINNER) return;
  int d = t & (DINNER-1);
  int m = t >> 9;
  float acc = bdt[d];
  #pragma unroll
  for (int r = 0; r < DTRANK; r++) acc += dbl[(size_t)m*NDBL + r] * wdt[d*DTRANK + r];
  delta[t] = softplusf_(acc);
}

// ------------------------------------------------- selective scan
// thread per (b, d, n): h = exp(delta*A)*h + delta*B*x ; y = sum_n h*C
// delta/x are uniform over the 16-lane state group: lane n==0 loads, shfl-broadcast.
__global__ void scan_kernel(const float* __restrict__ delta, const float* __restrict__ xc,
                            const float* __restrict__ dbl, const float* __restrict__ A_log,
                            float* __restrict__ y){
  int t = blockIdx.x*blockDim.x + threadIdx.x;  // 0 .. BATCH*DINNER*DSTATE-1
  int lane = threadIdx.x & 31;
  int gbase = lane & 16;                        // base lane of 16-lane state group
  int n = t & 15;
  int d = (t >> 4) & (DINNER-1);
  int b = t >> 13;
  float A = -__expf(A_log[d*DSTATE + n]);
  float h = 0.f;
  int mbase = b*SEQ;
  for (int l = 0; l < SEQ; ++l){
    int m = mbase + l;
    float dt = 0.f, xv = 0.f;
    if (n == 0){
      dt = delta[(size_t)m*DINNER + d];
      xv = xc[(size_t)m*DINNER + d];
    }
    dt = __shfl(dt, gbase, 32);
    xv = __shfl(xv, gbase, 32);
    float Bv = dbl[(size_t)m*NDBL + DTRANK + n];
    float Cv = dbl[(size_t)m*NDBL + DTRANK + DSTATE + n];
    h = __expf(dt*A)*h + dt*Bv*xv;
    float p = h*Cv;
    p += __shfl_xor(p, 1);
    p += __shfl_xor(p, 2);
    p += __shfl_xor(p, 4);
    p += __shfl_xor(p, 8);
    if (n == 0) y[(size_t)m*DINNER + d] = p;
  }
}

// ------------------------------------------------- y = (y + xc*D) * silu(z) -> bf16
__global__ void gate_kernel(const float* __restrict__ y, const float* __restrict__ xc,
                            const float* __restrict__ Dp, const float* __restrict__ xz,
                            __bf16* __restrict__ ybf){
  int t = blockIdx.x*blockDim.x + threadIdx.x;
  if (t >= NTOK*DINNER) return;
  int d = t & (DINNER-1);
  int m = t >> 9;
  float z = xz[(size_t)m*(2*DINNER) + DINNER + d];
  float v = (y[t] + xc[t]*Dp[d]) * (z * sigmoidf_(z));
  ybf[t] = (__bf16)v;
}

// =================================================================
extern "C" void kernel_launch(void* const* d_in, const int* in_sizes, int n_in,
                              void* d_out, int out_size, void* d_ws, size_t ws_size,
                              hipStream_t stream) {
  const float* x         = (const float*)d_in[0];
  const float* norm_w    = (const float*)d_in[1];
  const float* in_proj_w = (const float*)d_in[2];
  const float* conv_w    = (const float*)d_in[3];
  const float* conv_b    = (const float*)d_in[4];
  const float* x_proj_w  = (const float*)d_in[5];
  const float* dt_proj_w = (const float*)d_in[6];
  const float* dt_proj_b = (const float*)d_in[7];
  const float* A_log     = (const float*)d_in[8];
  const float* Dp        = (const float*)d_in[9];
  const float* out_proj_w= (const float*)d_in[10];
  float* out = (float*)d_out;

  // workspace carve-up (256B aligned)
  char* ws = (char*)d_ws;
  size_t off = 0;
  auto carve = [&](size_t bytes)->char*{ char* p = ws + off; off = (off + bytes + 255) & ~(size_t)255; return p; };
  __bf16* Winbf = (__bf16*)carve((size_t)2*DINNER*DMODEL*2);   // (1024,256) bf16
  __bf16* Xpbf  = (__bf16*)carve((size_t)NDBL*DINNER*2);       // (48,512)
  __bf16* Wobf  = (__bf16*)carve((size_t)DMODEL*DINNER*2);     // (256,512)
  __bf16* Hbf   = (__bf16*)carve((size_t)NTOK*DMODEL*2);       // normed tokens
  float*  XZ    = (float*)carve((size_t)NTOK*2*DINNER*4);      // in_proj output
  float*  XC    = (float*)carve((size_t)NTOK*DINNER*4);        // conv output f32
  __bf16* XCbf  = (__bf16*)carve((size_t)NTOK*DINNER*2);
  float*  DBL   = (float*)carve((size_t)NTOK*NDBL*4);          // x_proj output
  float*  DELTA = (float*)carve((size_t)NTOK*DINNER*4);
  float*  Yb    = (float*)carve((size_t)NTOK*DINNER*4);        // scan output
  __bf16* Ybf   = (__bf16*)carve((size_t)NTOK*DINNER*2);
  (void)ws_size; (void)in_sizes; (void)n_in; (void)out_size;

  // 0) weight conversions to bf16
  f2bf_kernel<<<(2*DINNER*DMODEL + 255)/256, 256, 0, stream>>>(in_proj_w, Winbf, 2*DINNER*DMODEL);
  f2bf_kernel<<<(NDBL*DINNER     + 255)/256, 256, 0, stream>>>(x_proj_w,  Xpbf,  NDBL*DINNER);
  f2bf_kernel<<<(DMODEL*DINNER   + 255)/256, 256, 0, stream>>>(out_proj_w,Wobf,  DMODEL*DINNER);

  // 1) RMSNorm + transpose -> Hbf (B*L, 256)
  rmsnorm_kernel<<<NTOK, DMODEL, 0, stream>>>(x, norm_w, Hbf);

  // 2) in_proj: XZ (4096 x 1024) = Hbf (4096x256) * Win^T   [32x64 tiles]
  {
    int ntiles = (2*DINNER)/64;                 // 16
    int total  = (NTOK/32) * ntiles;            // 128*16 = 2048
    gemm_bf16_t<2,4><<<(total + 7)/8, 256, 0, stream>>>(Hbf, Winbf, XZ,
        DMODEL, DMODEL, DMODEL, 2*DINNER, ntiles, total);
  }

  // 3) depthwise causal conv + SiLU
  conv_silu_kernel<<<(NTOK*DINNER + 255)/256, 256, 0, stream>>>(XZ, conv_w, conv_b, XC, XCbf);

  // 4) x_proj: DBL (4096 x 48) = XCbf (4096x512) * Wx^T   [32x48 tiles, no predication]
  {
    int ntiles = 1;                             // N = 48 = 16*3 exactly
    int total  = (NTOK/32) * ntiles;            // 128
    gemm_bf16_t<2,3><<<(total + 7)/8, 256, 0, stream>>>(XCbf, Xpbf, DBL,
        DINNER, DINNER, DINNER, NDBL, ntiles, total);
  }

  // 5) delta = softplus(dt_proj)
  dtproj_kernel<<<(NTOK*DINNER + 255)/256, 256, 0, stream>>>(DBL, dt_proj_w, dt_proj_b, DELTA);

  // 6) selective scan (sequential over L, parallel over B*Dinner*N = 16384 lanes)
  scan_kernel<<<(BATCH*DINNER*DSTATE + 255)/256, 256, 0, stream>>>(DELTA, XC, DBL, A_log, Yb);

  // 7) gate: y = (y + xc*D) * silu(z) -> bf16
  gate_kernel<<<(NTOK*DINNER + 255)/256, 256, 0, stream>>>(Yb, XC, Dp, XZ, Ybf);

  // 8) out_proj + residual + transpose back to (B, C, 1, L)  [32x64 tiles]
  {
    int total = (NTOK/32) * (DMODEL/64);        // 128*4 = 512
    gemm_out_residual_kernel<<<(total + 7)/8, 256, 0, stream>>>(Ybf, Wobf, x, out);
  }
}
